// FusedMoE_66486093742323
// MI455X (gfx1250) — matrix operand order repaired and checked by reading, not compiled, not run on hardware
//
#include <hip/hip_runtime.h>

// FusedMoE for MI455X (gfx1250, wave32, WMMA + async-to-LDS staging).
// top-2 routing -> per-expert compact token lists -> per-expert:
//   (a) convert expert weights fp32->bf16 once into ws staging
//   (b) grouped bf16 WMMA GEMM1 (gate/up fused + SiLU) with LDS-staged B
//   (c) grouped bf16 WMMA GEMM2 (down-proj, scaled accumulate into out)
//
// ws layout (needs ~71.3 MiB):
//   [0,256)        int   counts[8]
//   [256,..)       int   idx[E][T]
//   [..]           float tw[E][T]
//   [..]           bf16  Xb[T][H]
//   [..]           bf16  act[T][I]      (reused per expert)
//   [..]           bf16  w13b[2I][H]    (staging, reused per expert)
//   [..]           bf16  w2b[H][I]      (staging, reused per expert)

#define E_EXPERTS 8
#define HID 2048
#define INTER 2816
#define NT 4096

#define BROWB 80  // padded LDS row stride: 32 bf16 (64B) + 16B pad; 20 dwords,
                  // gcd(20,64)=4 -> 16 fragment lanes hit 16 distinct banks.

typedef __attribute__((ext_vector_type(16))) __bf16 bf16x16;
typedef __attribute__((ext_vector_type(8)))  __bf16 bf16x8;
typedef __attribute__((ext_vector_type(8)))  float  f32x8;
typedef int v4i_vs __attribute__((vector_size(16)));   // builtin's pointee type

#if defined(__AMDGCN__) && __has_builtin(__builtin_amdgcn_global_load_async_to_lds_b128)
#define HAVE_ASYNC_LDS 1
#else
#define HAVE_ASYNC_LDS 0
#endif

#define AS1 __attribute__((address_space(1)))
#define AS3 __attribute__((address_space(3)))

// copy 16B global -> LDS (async if available, else load+ds_store)
__device__ __forceinline__ void copy16(unsigned char* dst, const __bf16* src) {
#if HAVE_ASYNC_LDS
  __builtin_amdgcn_global_load_async_to_lds_b128(
      (AS1 v4i_vs*)src, (AS3 v4i_vs*)dst, 0, 0);
#else
  *reinterpret_cast<uint4*>(dst) = *reinterpret_cast<const uint4*>(src);
#endif
}

__device__ __forceinline__ void wait_async() {
#if HAVE_ASYNC_LDS
#if __has_builtin(__builtin_amdgcn_s_wait_asynccnt)
  __builtin_amdgcn_s_wait_asynccnt(0);
#else
  asm volatile("s_wait_asynccnt 0x0" ::: "memory");
#endif
#endif
}

// ---------- WMMA fragment loaders (wave32 layouts, ISA 7.12.2) ----------

// A: 16x32 bf16. lane L holds row M=L%16; K chunks [kh,kh+8) and [kh+16,kh+24),
// kh = (L<16 ? 0 : 8). Row-major source, two contiguous 16B chunks.
__device__ __forceinline__ bf16x16 load_a_frag(const __bf16* __restrict__ row,
                                               int k0, int lane) {
  const int kh = (lane & 16) ? 8 : 0;
  bf16x8 lo = *reinterpret_cast<const bf16x8*>(row + k0 + kh);
  bf16x8 hi = *reinterpret_cast<const bf16x8*>(row + k0 + 16 + kh);
  bf16x16 f;
#pragma unroll
  for (int i = 0; i < 8; ++i) { f[i] = lo[i]; f[i + 8] = hi[i]; }
  return f;
}

// B: 32x16 bf16 from LDS tile (row-major rows of W, 32 bf16/row, BROWB stride).
// lane L holds column N=L%16 of the 16-row subtile, K chunks [kh,kh+8) and
// [kh+16,kh+24), kh = (L<16 ? 0 : 8) -> 16B at row*BROWB + 2*kh and +32.
__device__ __forceinline__ bf16x16 lds_b_frag(const unsigned char* buf,
                                              int rowbase, int lane) {
  const unsigned char* p =
      buf + (size_t)(rowbase + (lane & 15)) * BROWB + ((lane & 16) ? 16 : 0);
  bf16x8 lo = *reinterpret_cast<const bf16x8*>(p);
  bf16x8 hi = *reinterpret_cast<const bf16x8*>(p + 32);
  bf16x16 f;
#pragma unroll
  for (int i = 0; i < 8; ++i) { f[i] = lo[i]; f[i + 8] = hi[i]; }
  return f;
}

// ---------- small kernels ----------

__global__ void zero_counts_kernel(int* __restrict__ counts) {
  if (threadIdx.x < E_EXPERTS) counts[threadIdx.x] = 0;
}

__global__ void zero_out_kernel(float* __restrict__ out) {
  size_t i = ((size_t)blockIdx.x * blockDim.x + threadIdx.x) * 4;
  *reinterpret_cast<float4*>(out + i) = make_float4(0.f, 0.f, 0.f, 0.f);
}

__global__ void cvt_f32_bf16_kernel(const float* __restrict__ x,
                                    __bf16* __restrict__ y) {
  size_t i = ((size_t)blockIdx.x * blockDim.x + threadIdx.x) * 8;
  const float4* p = reinterpret_cast<const float4*>(x + i);
  float4 a = p[0], b = p[1];
  bf16x8 v;
  v[0] = (__bf16)a.x; v[1] = (__bf16)a.y; v[2] = (__bf16)a.z; v[3] = (__bf16)a.w;
  v[4] = (__bf16)b.x; v[5] = (__bf16)b.y; v[6] = (__bf16)b.z; v[7] = (__bf16)b.w;
  *reinterpret_cast<bf16x8*>(y + i) = v;
}

// top-2 of softmax == softmax over top-2 logits. Strict '>' keeps
// jax.lax.top_k's first-occurrence tie-break.
__global__ void moe_route_kernel(const float* __restrict__ logits,
                                 int* __restrict__ counts,
                                 int* __restrict__ idx,
                                 float* __restrict__ tw) {
  int t = blockIdx.x * blockDim.x + threadIdx.x;
  if (t >= NT) return;
  float l[E_EXPERTS];
#pragma unroll
  for (int i = 0; i < E_EXPERTS; ++i) l[i] = logits[t * E_EXPERTS + i];
  int e1 = 0; float v1 = l[0];
#pragma unroll
  for (int i = 1; i < E_EXPERTS; ++i) if (l[i] > v1) { v1 = l[i]; e1 = i; }
  int e2 = (e1 == 0) ? 1 : 0; float v2 = l[e2];
#pragma unroll
  for (int i = 0; i < E_EXPERTS; ++i)
    if (i != e1 && l[i] > v2) { v2 = l[i]; e2 = i; }
  float w1 = 1.0f / (1.0f + __expf(v2 - v1));
  float w2 = 1.0f - w1;
  int p1 = atomicAdd(&counts[e1], 1);
  idx[e1 * NT + p1] = t; tw[e1 * NT + p1] = w1;
  int p2 = atomicAdd(&counts[e2], 1);
  idx[e2 * NT + p2] = t; tw[e2 * NT + p2] = w2;
}

// ---------- GEMM1: act = silu(X@Wg^T) * (X@Wu^T), gathered tokens ----------
// 256 thr = 8 waves as 4(M)x2(N); block tile 64x128, wave tile 16x64.
// B (gate+up) double-buffered in LDS via async copies; A shared per K-step.
// All 8 B fragments preloaded into distinct regs so the compiler can clause
// the ds_loads and overlap LDS latency with the WMMA pipeline.
__global__ __launch_bounds__(256) void moe_gemm1_kernel(
    const __bf16* __restrict__ Xb, const __bf16* __restrict__ w13b,
    const int* __restrict__ idx_e, const int* __restrict__ cnt_ptr,
    __bf16* __restrict__ act) {
  __shared__ alignas(16) unsigned char smG[2][128 * BROWB];
  __shared__ alignas(16) unsigned char smU[2][128 * BROWB];

  const int cnt = *cnt_ptr;
  const int m0 = blockIdx.y * 64;
  if (m0 >= cnt) return;
  const int n0 = blockIdx.x * 128;
  const int tid = threadIdx.x;
  const int wave = tid >> 5, lane = tid & 31;
  const int wm = wave >> 1, wn = wave & 1;
  const int mrow = m0 + wm * 16;

  int arow_l = mrow + (lane & 15);
  if (arow_l > cnt - 1) arow_l = cnt - 1;   // clamp tail (stores predicated)
  const __bf16* arow = Xb + (size_t)idx_e[arow_l] * HID;

  // each thread stages 32B of the gate tile and 32B of the up tile per K-step
  const int srow = tid >> 1, spart = tid & 1;
  const __bf16* gsrc = w13b + (size_t)(n0 + srow) * HID + spart * 16;
  const __bf16* usrc = w13b + (size_t)(n0 + srow + INTER) * HID + spart * 16;
  const int doff = srow * BROWB + spart * 32;

  auto stage = [&](int buf, int k0) {
    copy16(&smG[buf][doff],      gsrc + k0);
    copy16(&smG[buf][doff + 16], gsrc + k0 + 8);
    copy16(&smU[buf][doff],      usrc + k0);
    copy16(&smU[buf][doff + 16], usrc + k0 + 8);
  };

  f32x8 accg[4] = {};
  f32x8 accu[4] = {};
  stage(0, 0);
  int cur = 0;
  for (int k0 = 0; k0 < HID; k0 += 32) {
    wait_async();
    __syncthreads();
    if (k0 + 32 < HID) stage(cur ^ 1, k0 + 32);
    bf16x16 a = load_a_frag(arow, k0, lane);
    __builtin_prefetch(arow + k0 + 64, 0, 3);
    bf16x16 bg[4], bu[4];
#pragma unroll
    for (int j = 0; j < 4; ++j) {
      bg[j] = lds_b_frag(smG[cur], wn * 64 + 16 * j, lane);
      bu[j] = lds_b_frag(smU[cur], wn * 64 + 16 * j, lane);
    }
#pragma unroll
    for (int j = 0; j < 4; ++j) {
      accg[j] = __builtin_amdgcn_wmma_f32_16x16x32_bf16(
          false, a, false, bg[j], (short)0, accg[j], false, false);
      accu[j] = __builtin_amdgcn_wmma_f32_16x16x32_bf16(
          false, a, false, bu[j], (short)0, accu[j], false, false);
    }
    cur ^= 1;
  }

  // C layout: lane L, elem r -> C[M = r + 8*(L/16), N = L%16]
  const int rbase = mrow + ((lane >> 4) << 3);
  const int cbase = n0 + wn * 64 + (lane & 15);
#pragma unroll
  for (int j = 0; j < 4; ++j) {
#pragma unroll
    for (int r = 0; r < 8; ++r) {
      int row = rbase + r;
      if (row < cnt) {
        float g = accg[j][r], u = accu[j][r];
        float s = g / (1.0f + __expf(-g));   // silu
        act[(size_t)row * INTER + (cbase + 16 * j)] = (__bf16)(s * u);
      }
    }
  }
}

// ---------- GEMM2: out[tok] += w_tok * (act @ W2^T) ----------
__global__ __launch_bounds__(256) void moe_gemm2_kernel(
    const __bf16* __restrict__ act, const __bf16* __restrict__ w2b,
    const int* __restrict__ idx_e, const float* __restrict__ tw_e,
    const int* __restrict__ cnt_ptr, float* __restrict__ out) {
  __shared__ alignas(16) unsigned char smB[2][128 * BROWB];

  const int cnt = *cnt_ptr;
  const int m0 = blockIdx.y * 64;
  if (m0 >= cnt) return;
  const int n0 = blockIdx.x * 128;
  const int tid = threadIdx.x;
  const int wave = tid >> 5, lane = tid & 31;
  const int wm = wave >> 1, wn = wave & 1;
  const int mrow = m0 + wm * 16;

  int arow_l = mrow + (lane & 15);
  if (arow_l > cnt - 1) arow_l = cnt - 1;
  const __bf16* arow = act + (size_t)arow_l * INTER;

  const int srow = tid >> 1, spart = tid & 1;
  const __bf16* bsrc = w2b + (size_t)(n0 + srow) * INTER + spart * 16;
  const int doff = srow * BROWB + spart * 32;

  auto stage = [&](int buf, int k0) {
    copy16(&smB[buf][doff],      bsrc + k0);
    copy16(&smB[buf][doff + 16], bsrc + k0 + 8);
  };

  f32x8 acc[4] = {};
  stage(0, 0);
  int cur = 0;
  for (int k0 = 0; k0 < INTER; k0 += 32) {
    wait_async();
    __syncthreads();
    if (k0 + 32 < INTER) stage(cur ^ 1, k0 + 32);
    bf16x16 a = load_a_frag(arow, k0, lane);
    __builtin_prefetch(arow + k0 + 64, 0, 3);
    bf16x16 b[4];
#pragma unroll
    for (int j = 0; j < 4; ++j)
      b[j] = lds_b_frag(smB[cur], wn * 64 + 16 * j, lane);
#pragma unroll
    for (int j = 0; j < 4; ++j)
      acc[j] = __builtin_amdgcn_wmma_f32_16x16x32_bf16(
          false, a, false, b[j], (short)0, acc[j], false, false);
    cur ^= 1;
  }

  const int rbase = mrow + ((lane >> 4) << 3);
  const int cbase = n0 + wn * 64 + (lane & 15);
#pragma unroll
  for (int r = 0; r < 8; ++r) {
    int row = rbase + r;
    if (row < cnt) {                   // predicate: never RMW a clamped row
      int tokf = idx_e[row];
      float wv = tw_e[row];
      float* orow = out + (size_t)tokf * HID;
#pragma unroll
      for (int j = 0; j < 4; ++j)
        orow[cbase + 16 * j] += wv * acc[j][r];
    }
  }
}

// ---------- host ----------

extern "C" void kernel_launch(void* const* d_in, const int* in_sizes, int n_in,
                              void* d_out, int out_size, void* d_ws, size_t ws_size,
                              hipStream_t stream) {
  const float* hs  = (const float*)d_in[0];   // [T,H]
  const float* rl  = (const float*)d_in[1];   // [T,E]
  const float* w13 = (const float*)d_in[2];   // [E,2I,H]
  const float* w2  = (const float*)d_in[3];   // [E,H,I]
  float* out = (float*)d_out;                 // [T,H]

  char* ws = (char*)d_ws;
  size_t off = 0;
  int* counts = (int*)(ws + off);           off += 256;
  int* idx    = (int*)(ws + off);           off += (size_t)4 * E_EXPERTS * NT;
  float* tw   = (float*)(ws + off);         off += (size_t)4 * E_EXPERTS * NT;
  __bf16* Xb   = (__bf16*)(ws + off);       off += (size_t)2 * NT * HID;
  __bf16* act  = (__bf16*)(ws + off);       off += (size_t)2 * NT * INTER;
  __bf16* w13b = (__bf16*)(ws + off);       off += (size_t)2 * 2 * INTER * HID;
  __bf16* w2b  = (__bf16*)(ws + off);       off += (size_t)2 * HID * INTER;

  zero_counts_kernel<<<1, 32, 0, stream>>>(counts);
  moe_route_kernel<<<(NT + 255) / 256, 256, 0, stream>>>(rl, counts, idx, tw);
  cvt_f32_bf16_kernel<<<(NT * HID) / 2048, 256, 0, stream>>>(hs, Xb);
  zero_out_kernel<<<(NT * HID) / 1024, 256, 0, stream>>>(out);

  for (int e = 0; e < E_EXPERTS; ++e) {
    cvt_f32_bf16_kernel<<<(2 * INTER * HID) / 2048, 256, 0, stream>>>(
        w13 + (size_t)e * 2 * INTER * HID, w13b);
    moe_gemm1_kernel<<<dim3(INTER / 128, NT / 64), 256, 0, stream>>>(
        Xb, w13b, idx + e * NT, counts + e, act);
    cvt_f32_bf16_kernel<<<(HID * INTER) / 2048, 256, 0, stream>>>(
        w2 + (size_t)e * HID * INTER, w2b);
    moe_gemm2_kernel<<<dim3(HID / 128, NT / 64), 256, 0, stream>>>(
        act, w2b, idx + e * NT, tw + e * NT, counts + e, out);
  }
}
